// ABAgInteractionLayer_58265526338055
// MI455X (gfx1250) — compile-verified, matrix operand order checked by reading, not run on hardware
//
#include <hip/hip_runtime.h>
#include <cstdint>

#define B_  8
#define LQ  2560   // 512*5
#define LK  5120   // 1024*5
#define DD  256

typedef __attribute__((ext_vector_type(16))) __bf16 v16bf;
typedef __attribute__((ext_vector_type(8)))  __bf16 v8bf;
typedef __attribute__((ext_vector_type(8)))  float  v8f;

static __device__ __forceinline__ __bf16 f2bf(float f) {
  union { float f; uint32_t u; } in; in.f = f;
  uint32_t r = (in.u + 0x7FFFu + ((in.u >> 16) & 1u)) >> 16;
  union { uint16_t u; __bf16 h; } out; out.u = (uint16_t)r;
  return out.h;
}

static __device__ __forceinline__ v16bf mk16(v8bf lo, v8bf hi) {
  v16bf r;
#pragma unroll
  for (int i = 0; i < 8; ++i) { r[i] = lo[i]; r[i + 8] = hi[i]; }
  return r;
}

// A-operand (and mirror-B-operand) fragment load, bf16 source.
// Element e in [0,8)  -> K = half*8 + e        (contiguous 16B run)
// Element e in [8,16) -> K = 16 + half*8 + e-8 (contiguous 16B run)
static __device__ __forceinline__ v16bf load_frag_bf16(const __bf16* base, int stride,
                                                       int mrow, int half) {
  const __bf16* p = base + (size_t)mrow * stride + half * 8;
  v8bf lo = *(const v8bf*)p;
  v8bf hi = *(const v8bf*)(p + 16);
  return mk16(lo, hi);
}

// Same fragment, f32 source converted to bf16 on the fly.
static __device__ __forceinline__ v16bf load_frag_f32(const float* base, int stride,
                                                      int mrow, int half) {
  const float* p = base + (size_t)mrow * stride + half * 8;
  v16bf r;
#pragma unroll
  for (int i = 0; i < 8; ++i) r[i] = f2bf(p[i]);
#pragma unroll
  for (int i = 0; i < 8; ++i) r[i + 8] = f2bf(p[16 + i]);
  return r;
}

static __device__ __forceinline__ v8f wmma_bf16(v16bf a, v16bf b, v8f c) {
  return __builtin_amdgcn_wmma_f32_16x16x32_bf16(false, a, false, b, (short)0, c,
                                                 false, false);
}

// Async global -> LDS, 128-bit per lane, tracked by ASYNCcnt.
// VDST operand = per-lane LDS byte address (low 32 bits of generic LDS ptr).
static __device__ __forceinline__ void async_b128(uint32_t lds_addr, const void* gaddr) {
  asm volatile("global_load_async_to_lds_b128 %0, %1, off"
               :: "v"(lds_addr), "v"(gaddr) : "memory");
}

// ---------------------------------------------------------------------------
// Weight transpose+convert: Wt[n][k] = bf16(W[k][n]), 256x256
__global__ __launch_bounds__(256) void wtr_k(const float* __restrict__ W,
                                             __bf16* __restrict__ Wt) {
  int t = blockIdx.x * 256 + threadIdx.x;
  int n = t >> 8, k = t & 255;
  Wt[(size_t)n * 256 + k] = f2bf(W[(size_t)k * 256 + n]);
}

// ---------------------------------------------------------------------------
// Row masks: one wave per 256-feature row.
// mode 0: out = valid ? 1 : 0      (query mask, multiplicative)
// mode 1: out = valid ? 0 : -1e9   (key mask, additive score bias)
__global__ __launch_bounds__(256) void mask_k(const float* __restrict__ X,
                                              float* __restrict__ out,
                                              int rows, int mode) {
  int row = blockIdx.x * 8 + (threadIdx.x >> 5);
  int lane = threadIdx.x & 31;
  if (row >= rows) return;
  const float* p = X + (size_t)row * 256 + lane * 8;
  float s = 0.f;
#pragma unroll
  for (int i = 0; i < 8; ++i) s += fabsf(p[i]);
#pragma unroll
  for (int off = 16; off; off >>= 1) s += __shfl_xor(s, off, 32);
  if (lane == 0)
    out[row] = (mode == 0) ? (s > 0.f ? 1.f : 0.f) : (s > 0.f ? 0.f : -1e9f);
}

// ---------------------------------------------------------------------------
// GEMM: Y = X(f32)[M,256] @ W[256,256] + bias, via bf16 WMMA.
// Wt is the pre-transposed bf16 weight (Wt[n][k] = W[k][n]).
// MODE 0: Ybf[m*256+n]  (row-major bf16)
// MODE 1: Ybf[b*256*Lkp + n*Lkp + l]  (per-batch transposed bf16; m = b*Lkp+l)
// MODE 2: Yf[m*256+n] = resid[m*256+n] + val  (f32, residual fused)
template <int MODE>
__global__ __launch_bounds__(256) void gemm_k(const float* __restrict__ X,
                                              const __bf16* __restrict__ Wt,
                                              const float* __restrict__ bias,
                                              int M,
                                              __bf16* __restrict__ Ybf,
                                              float* __restrict__ Yf,
                                              const float* __restrict__ resid,
                                              int Lkp) {
  const int wave = threadIdx.x >> 5, lane = threadIdx.x & 31;
  const int mrow = lane & 15, half = lane >> 4;
  const int m0 = (blockIdx.x * 8 + wave) * 16;
  if (m0 >= M) return;  // wave-uniform

  v8f c[16];
#pragma unroll
  for (int nt = 0; nt < 16; ++nt) {
    float bv = bias[nt * 16 + mrow];
#pragma unroll
    for (int r = 0; r < 8; ++r) c[nt][r] = bv;
  }

  for (int k0 = 0; k0 < 256; k0 += 32) {
    v16bf a = load_frag_f32(X + (size_t)m0 * 256 + k0, 256, mrow, half);
#pragma unroll
    for (int nt = 0; nt < 16; ++nt) {
      v16bf w = load_frag_bf16(Wt + (size_t)nt * 16 * 256 + k0, 256, mrow, half);
      c[nt] = wmma_bf16(a, w, c[nt]);
    }
  }

#pragma unroll
  for (int nt = 0; nt < 16; ++nt) {
#pragma unroll
    for (int r = 0; r < 8; ++r) {
      int mg = m0 + half * 8 + r;
      int n = nt * 16 + mrow;
      float v = c[nt][r];
      if (MODE == 0) {
        Ybf[(size_t)mg * 256 + n] = f2bf(v);
      } else if (MODE == 1) {
        int bb = mg / Lkp, l = mg - bb * Lkp;
        Ybf[(size_t)bb * 256 * Lkp + (size_t)n * Lkp + l] = f2bf(v);
      } else {
        size_t idx = (size_t)mg * 256 + n;
        Yf[idx] = resid[idx] + v;
      }
    }
  }
}

// ---------------------------------------------------------------------------
// Flash attention: grid (B, LQ/64), block 128 (4 waves; each wave = 16 q-rows).
// KV tiles of 32 double-buffered in LDS via async global->LDS loads (ASYNCcnt),
// so the tile-(j+1) fetch overlaps the 32 WMMAs of tile j.
__global__ __launch_bounds__(128) void attn_k(const __bf16* __restrict__ Qb,
                                              const __bf16* __restrict__ Kb,
                                              const __bf16* __restrict__ Vt,
                                              const float* __restrict__ agb,
                                              const float* __restrict__ abm,
                                              float* __restrict__ Obuf) {
  __shared__ __align__(16) __bf16 sK[2][32 * 256];   // K tile, row-major [kv][d]
  __shared__ __align__(16) __bf16 sV[2][256 * 32];   // V tile, d-major  [d][kv]
  __shared__ __align__(16) __bf16 sP[4][16 * 32];    // per-wave P staging

  const int b = blockIdx.x;
  const int wave = threadIdx.x >> 5, lane = threadIdx.x & 31;
  const int mrow = lane & 15, half = lane >> 4;
  const int q0 = blockIdx.y * 64 + wave * 16;

  // Issue one KV tile's async loads: 16 x b128 per lane-slot (8 K + 8 V) per wave.
  auto issue_tile = [&](int j0n, int bufn) {
    const __bf16* kbase = Kb + ((size_t)b * LK + j0n) * DD;
    const __bf16* vbase = Vt + (size_t)b * DD * LK + j0n;
#pragma unroll
    for (int i = 0; i < 8; ++i) {
      int unit = threadIdx.x + i * 128;
      int jr = unit >> 5, u = unit & 31;
      async_b128((uint32_t)(uintptr_t)(&sK[bufn][jr * 256 + u * 8]),
                 kbase + (size_t)jr * DD + u * 8);
    }
#pragma unroll
    for (int i = 0; i < 8; ++i) {
      int unit = threadIdx.x + i * 128;
      int d = unit >> 2, u = unit & 3;
      async_b128((uint32_t)(uintptr_t)(&sV[bufn][d * 32 + u * 8]),
                 vbase + (size_t)d * LK + u * 8);
    }
  };

  // Q fragments for this wave's 16 rows, full 256-deep (8 k-chunks), kept in VGPRs.
  v16bf qf[8];
#pragma unroll
  for (int kk = 0; kk < 8; ++kk)
    qf[kk] = load_frag_bf16(Qb + ((size_t)b * LQ + q0) * DD + kk * 32, DD, mrow, half);

  v8f o[16];
#pragma unroll
  for (int nt = 0; nt < 16; ++nt)
#pragma unroll
    for (int r = 0; r < 8; ++r) o[nt][r] = 0.f;

  float mx[8], sl[8];
#pragma unroll
  for (int r = 0; r < 8; ++r) { mx[r] = -1e30f; sl[r] = 0.f; }

  issue_tile(0, 0);  // prologue: tile 0 in flight

  for (int j0 = 0, it = 0; j0 < LK; j0 += 32, ++it) {
    const int cur = it & 1;
    if (j0 + 32 < LK) {
      issue_tile(j0 + 32, cur ^ 1);
      // 16 newest in flight for tile j+1; wait until only those remain -> tile j done.
      asm volatile("s_wait_asynccnt 0x10" ::: "memory");
    } else {
      asm volatile("s_wait_asynccnt 0x0" ::: "memory");
    }
    __syncthreads();  // everyone's tile-j data visible in LDS

    const __bf16* kbuf = sK[cur];
    const __bf16* vbuf = sV[cur];

    // S = Q @ K^T for two 16-wide kv sub-tiles (K rows loaded as mirror-B).
    v8f s0 = {}, s1 = {};
#pragma unroll
    for (int kk = 0; kk < 8; ++kk) {
      v16bf f0 = load_frag_bf16(kbuf + kk * 32, 256, mrow, half);
      s0 = wmma_bf16(qf[kk], f0, s0);
      v16bf f1 = load_frag_bf16(kbuf + 16 * 256 + kk * 32, 256, mrow, half);
      s1 = wmma_bf16(qf[kk], f1, s1);
    }

    const float scale = 0.0625f;  // 1/sqrt(256)
    float bias0 = agb[(size_t)b * LK + j0 + mrow];
    float bias1 = agb[(size_t)b * LK + j0 + 16 + mrow];
#pragma unroll
    for (int r = 0; r < 8; ++r) {
      s0[r] = s0[r] * scale + bias0;
      s1[r] = s1[r] * scale + bias1;
    }

    // Online softmax: row reductions across the 16 lanes of each half.
    float tm[8];
#pragma unroll
    for (int r = 0; r < 8; ++r) tm[r] = fmaxf(s0[r], s1[r]);
#pragma unroll
    for (int off = 8; off; off >>= 1)
#pragma unroll
      for (int r = 0; r < 8; ++r) tm[r] = fmaxf(tm[r], __shfl_xor(tm[r], off, 32));

    float corr[8];
#pragma unroll
    for (int r = 0; r < 8; ++r) {
      float mn = fmaxf(mx[r], tm[r]);
      corr[r] = __expf(mx[r] - mn);
      mx[r] = mn;
    }
#pragma unroll
    for (int r = 0; r < 8; ++r) {
      s0[r] = __expf(s0[r] - mx[r]);
      s1[r] = __expf(s1[r] - mx[r]);
    }
    float ts[8];
#pragma unroll
    for (int r = 0; r < 8; ++r) ts[r] = s0[r] + s1[r];
#pragma unroll
    for (int off = 8; off; off >>= 1)
#pragma unroll
      for (int r = 0; r < 8; ++r) ts[r] += __shfl_xor(ts[r], off, 32);
#pragma unroll
    for (int r = 0; r < 8; ++r) sl[r] = sl[r] * corr[r] + ts[r];

#pragma unroll
    for (int nt = 0; nt < 16; ++nt)
#pragma unroll
      for (int r = 0; r < 8; ++r) o[nt][r] *= corr[r];

    // C-layout P -> row-major LDS patch (wave-private; DS ops in-order per wave).
    __bf16* pw = sP[wave];
#pragma unroll
    for (int r = 0; r < 8; ++r) {
      pw[(half * 8 + r) * 32 + mrow] = f2bf(s0[r]);
      pw[(half * 8 + r) * 32 + 16 + mrow] = f2bf(s1[r]);
    }
    v16bf pf = load_frag_bf16(pw, 32, mrow, half);

    // O += P @ V  (V rows from transposed tile act as mirror-B).
#pragma unroll
    for (int nt = 0; nt < 16; ++nt) {
      v16bf vf = load_frag_bf16(vbuf + nt * 16 * 32, 32, mrow, half);
      o[nt] = wmma_bf16(pf, vf, o[nt]);
    }
    __syncthreads();  // tile-j buffer free before iter j+1 overwrites it
  }

  float inv[8];
#pragma unroll
  for (int r = 0; r < 8; ++r) {
    int qg = q0 + half * 8 + r;
    inv[r] = abm[(size_t)b * LQ + qg] / sl[r];
  }
#pragma unroll
  for (int nt = 0; nt < 16; ++nt)
#pragma unroll
    for (int r = 0; r < 8; ++r) {
      int qg = q0 + half * 8 + r;
      Obuf[((size_t)b * LQ + qg) * DD + nt * 16 + mrow] = o[nt][r] * inv[r];
    }
}

// ---------------------------------------------------------------------------
extern "C" void kernel_launch(void* const* d_in, const int* in_sizes, int n_in,
                              void* d_out, int out_size, void* d_ws, size_t ws_size,
                              hipStream_t stream) {
  (void)in_sizes; (void)n_in; (void)out_size; (void)ws_size;

  const float* ab = (const float*)d_in[0];
  const float* ag = (const float*)d_in[1];
  const float* Wq = (const float*)d_in[2];
  const float* bq = (const float*)d_in[3];
  const float* Wk = (const float*)d_in[4];
  const float* bk = (const float*)d_in[5];
  const float* Wv = (const float*)d_in[6];
  const float* bv = (const float*)d_in[7];
  const float* Wo = (const float*)d_in[8];
  const float* bo = (const float*)d_in[9];
  float* out = (float*)d_out;

  char* ws = (char*)d_ws;
  size_t off = 0;
  auto alloc = [&](size_t bytes) -> char* {
    char* p = ws + off;
    off += (bytes + 255) & ~(size_t)255;
    return p;
  };
  __bf16* Wtq = (__bf16*)alloc((size_t)256 * 256 * 2);
  __bf16* Wtk = (__bf16*)alloc((size_t)256 * 256 * 2);
  __bf16* Wtv = (__bf16*)alloc((size_t)256 * 256 * 2);
  __bf16* Wto = (__bf16*)alloc((size_t)256 * 256 * 2);
  __bf16* Qb  = (__bf16*)alloc((size_t)B_ * LQ * DD * 2);
  __bf16* Kb  = (__bf16*)alloc((size_t)B_ * LK * DD * 2);
  __bf16* Vtb = (__bf16*)alloc((size_t)B_ * DD * LK * 2);
  float*  Obuf = (float*)alloc((size_t)B_ * LQ * DD * 4);
  float*  abm  = (float*)alloc((size_t)B_ * LQ * 4);
  float*  agb  = (float*)alloc((size_t)B_ * LK * 4);

  wtr_k<<<256, 256, 0, stream>>>(Wq, Wtq);
  wtr_k<<<256, 256, 0, stream>>>(Wk, Wtk);
  wtr_k<<<256, 256, 0, stream>>>(Wv, Wtv);
  wtr_k<<<256, 256, 0, stream>>>(Wo, Wto);

  mask_k<<<(B_ * LQ) / 8, 256, 0, stream>>>(ab, abm, B_ * LQ, 0);
  mask_k<<<(B_ * LK) / 8, 256, 0, stream>>>(ag, agb, B_ * LK, 1);

  gemm_k<0><<<(B_ * LQ) / 128, 256, 0, stream>>>(ab, Wtq, bq, B_ * LQ, Qb, nullptr, nullptr, 0);
  gemm_k<0><<<(B_ * LK) / 128, 256, 0, stream>>>(ag, Wtk, bk, B_ * LK, Kb, nullptr, nullptr, 0);
  gemm_k<1><<<(B_ * LK) / 128, 256, 0, stream>>>(ag, Wtv, bv, B_ * LK, Vtb, nullptr, nullptr, LK);

  attn_k<<<dim3(B_, LQ / 64), 128, 0, stream>>>(Qb, Kb, Vtb, agb, abm, Obuf);

  gemm_k<2><<<(B_ * LQ) / 128, 256, 0, stream>>>(Obuf, Wto, bo, B_ * LQ, nullptr, out, ab, 0);
}
